// CausalMultiHeadSelfAttention_54855322305199
// MI455X (gfx1250) — matrix operand order repaired and checked by reading, not compiled, not run on hardware
//
#include <hip/hip_runtime.h>

// ---------------------------------------------------------------------------
// Causal MHA for gfx1250 (MI455X): all GEMM paths via v_wmma_f32_16x16x32_bf16,
// double-buffered GLOBAL_LOAD_ASYNC_TO_LDS staging (ASYNCcnt-pipelined).
// ---------------------------------------------------------------------------

typedef __bf16 bf16_t;
typedef __attribute__((ext_vector_type(16))) __bf16 v16bf;
typedef __attribute__((ext_vector_type(8)))  float  v8f;
typedef __attribute__((ext_vector_type(4)))  int    v4i;

#define WMMA_BF16(A_, B_, C_) \
  __builtin_amdgcn_wmma_f32_16x16x32_bf16(false, (A_), false, (B_), (short)0, (C_), false, false)

constexpr int DMODEL = 2048;
constexpr int SEQ    = 2048;
constexpr int BATCH  = 4;
constexpr int HEADS  = 16;
constexpr int DHEAD  = 128;

#if defined(__has_builtin)
#if __has_builtin(__builtin_amdgcn_global_load_async_to_lds_b128)
#define ASYNC_CP 1
#endif
#if __has_builtin(__builtin_amdgcn_s_wait_asynccnt)
#define ASYNC_WAIT_BUILTIN 1
#endif
#endif

typedef __attribute__((address_space(1))) v4i gv4i;  // global (AS1) int4 vector
typedef __attribute__((address_space(3))) v4i lv4i;  // LDS    (AS3) int4 vector

// 16-byte global->LDS copy; async (ASYNCcnt-tracked, no VGPR staging) if available.
__device__ __forceinline__ void cp16(bf16_t* dst_lds, const bf16_t* src_g) {
#if defined(ASYNC_CP)
  __builtin_amdgcn_global_load_async_to_lds_b128((gv4i*)src_g, (lv4i*)dst_lds, 0, 0);
#else
  *(uint4*)dst_lds = *(const uint4*)src_g;
#endif
}

__device__ __forceinline__ void cp_wait_all() {
#if defined(ASYNC_CP)
#if defined(ASYNC_WAIT_BUILTIN)
  __builtin_amdgcn_s_wait_asynccnt(0);
#else
  asm volatile("s_wait_asynccnt 0x0" ::: "memory");
#endif
#endif
}

union FragU { uint4 u[2]; v16bf v; };

// A fragment (16x32, 16-bit): lane holds row (lane&15); K pairs at base and
// base+16 halves, base = (lane&16)?8:0.  Two 16B loads.
__device__ __forceinline__ v16bf load_frag_a(const bf16_t* base, int lda) {
  const int lane = threadIdx.x & 31;
  const bf16_t* p = base + (size_t)(lane & 15) * lda + ((lane & 16) ? 8 : 0);
  FragU f;
  f.u[0] = *(const uint4*)(p);
  f.u[1] = *(const uint4*)(p + 16);
  return f.v;
}

// B fragment (32x16, 16-bit): lane holds column (lane&15); K pairs at base and
// base+16 halves, base = (lane&16)?8:0.  Source is N-major [N][K].
__device__ __forceinline__ v16bf load_frag_b(const bf16_t* base, int ldb) {
  const int lane = threadIdx.x & 31;
  const bf16_t* p = base + (size_t)(lane & 15) * ldb + ((lane & 16) ? 8 : 0);
  FragU f;
  f.u[0] = *(const uint4*)(p);
  f.u[1] = *(const uint4*)(p + 16);
  return f.v;
}

// ---------------------------------------------------------------------------
// Stage 0a: f32 -> bf16 pack (x)
// ---------------------------------------------------------------------------
__global__ void pack_bf16_kernel(const float* __restrict__ x,
                                 bf16_t* __restrict__ xb, int n) {
  int i = (blockIdx.x * 256 + threadIdx.x) * 4;
  if (i + 3 < n) {
    float4 v = *(const float4*)(x + i);
    xb[i + 0] = (bf16_t)v.x;
    xb[i + 1] = (bf16_t)v.y;
    xb[i + 2] = (bf16_t)v.z;
    xb[i + 3] = (bf16_t)v.w;
  }
}

// ---------------------------------------------------------------------------
// Stage 0b: W[K][N] f32 -> Wt[N][K] bf16 (LDS-tiled transpose)
// ---------------------------------------------------------------------------
__global__ void transpose_to_bf16_kernel(const float* __restrict__ W,
                                         bf16_t* __restrict__ Wt) {
  __shared__ float tile[32][33];
  const int bx = blockIdx.x * 32;  // n
  const int by = blockIdx.y * 32;  // k
  const int tx = threadIdx.x, ty = threadIdx.y;
#pragma unroll
  for (int r = 0; r < 4; ++r)
    tile[ty + r * 8][tx] = W[(size_t)(by + ty + r * 8) * DMODEL + bx + tx];
  __syncthreads();
#pragma unroll
  for (int r = 0; r < 4; ++r)
    Wt[(size_t)(bx + ty + r * 8) * DMODEL + by + tx] = (bf16_t)tile[tx][ty + r * 8];
}

// ---------------------------------------------------------------------------
// Stage 1 & 3: GEMM  C[M=8192][N=2048] = A[M][K=2048] * Wt[N][K]^T
// Block tile 128x64, K-step 64, 256 threads = 8 waves (4x2), wave tile 32x32.
// Double-buffered async LDS staging: prefetch tile t+1 while computing t.
// MODE 0: bf16 row-major (Q,K)  MODE 1: bf16 Vt[b][h][d][s]  MODE 2: f32
// ---------------------------------------------------------------------------
template <int MODE>
__global__ void gemm_bf16_kernel(const bf16_t* __restrict__ A,
                                 const bf16_t* __restrict__ Bt,
                                 void* __restrict__ Cv) {
  constexpr int K = DMODEL, N = DMODEL;
  __shared__ bf16_t As[2][128][72];  // 2 x 18 KB (+8 halves pad)
  __shared__ bf16_t Bs[2][64][72];   // 2 x  9 KB

  const int tid  = threadIdx.x;
  const int lane = tid & 31;
  const int wave = tid >> 5;
  const int wm   = (wave & 3) * 32;
  const int wn   = (wave >> 2) * 32;
  const int m0   = blockIdx.y * 128;
  const int n0   = blockIdx.x * 64;

  // cooperative tile staging: A 1024 chunks (4/thread), B 512 chunks (2/thread)
  auto stage = [&](int buf, int k0) {
#pragma unroll
    for (int i = 0; i < 4; ++i) {
      int c = tid + 256 * i;
      int row = c >> 3, off = (c & 7) * 8;
      cp16(&As[buf][row][off], A + (size_t)(m0 + row) * K + k0 + off);
    }
#pragma unroll
    for (int i = 0; i < 2; ++i) {
      int c = tid + 256 * i;
      int row = c >> 3, off = (c & 7) * 8;
      cp16(&Bs[buf][row][off], Bt + (size_t)(n0 + row) * K + k0 + off);
    }
  };

  const v8f vz = {0.f, 0.f, 0.f, 0.f, 0.f, 0.f, 0.f, 0.f};
  v8f acc[2][2] = {{vz, vz}, {vz, vz}};

  stage(0, 0);
  cp_wait_all();
  __syncthreads();

  constexpr int NIT = K / 64;
  for (int it = 0; it < NIT; ++it) {
    const int buf = it & 1;
    if (it + 1 < NIT) stage(buf ^ 1, (it + 1) * 64);  // async prefetch
#pragma unroll
    for (int kk = 0; kk < 64; kk += 32) {
      v16bf a0 = load_frag_a(&As[buf][wm][kk], 72);
      v16bf a1 = load_frag_a(&As[buf][wm + 16][kk], 72);
      v16bf b0 = load_frag_b(&Bs[buf][wn][kk], 72);
      v16bf b1 = load_frag_b(&Bs[buf][wn + 16][kk], 72);
      acc[0][0] = WMMA_BF16(a0, b0, acc[0][0]);
      acc[0][1] = WMMA_BF16(a0, b1, acc[0][1]);
      acc[1][0] = WMMA_BF16(a1, b0, acc[1][0]);
      acc[1][1] = WMMA_BF16(a1, b1, acc[1][1]);
    }
    cp_wait_all();    // prefetch landed
    __syncthreads();  // everyone done reading buf before it is reused
  }

  const int cl = lane & 15;
  const int hi = (lane & 16) ? 8 : 0;
#pragma unroll
  for (int i = 0; i < 2; ++i) {
#pragma unroll
    for (int j = 0; j < 2; ++j) {
#pragma unroll
      for (int r = 0; r < 8; ++r) {
        const int row = m0 + wm + i * 16 + r + hi;
        const int col = n0 + wn + j * 16 + cl;
        const float v = acc[i][j][r];
        if (MODE == 0) {
          ((bf16_t*)Cv)[(size_t)row * N + col] = (bf16_t)v;
        } else if (MODE == 2) {
          ((float*)Cv)[(size_t)row * N + col] = v;
        } else {  // MODE 1: Vt[b][h][d][s]
          const int bi = row >> 11, s = row & (SEQ - 1);
          const int h = col >> 7, d = col & (DHEAD - 1);
          ((bf16_t*)Cv)[((size_t)((bi * HEADS + h) * DHEAD + d)) * SEQ + s] = (bf16_t)v;
        }
      }
    }
  }
}

// ---------------------------------------------------------------------------
// Stage 2: flash attention.  Block = 4 waves, 64 q-rows (16/wave), Dh=128.
// Double-buffered async K/V staging; Q fragments in wave-private LDS slab;
// online softmax; P re-shaped C-layout -> A-layout through per-wave LDS tile.
// ---------------------------------------------------------------------------
__global__ void flash_attn_kernel(const bf16_t* __restrict__ Q,
                                  const bf16_t* __restrict__ Kb,
                                  const bf16_t* __restrict__ Vt,
                                  bf16_t* __restrict__ attn) {
  __shared__ uint4  Qs[4][4][2][32];    // [wave][chunk][half][lane]    16 KB
  __shared__ bf16_t Pl[4][16][72];      // per-wave P tiles              9 KB
  __shared__ bf16_t Ks[2][64][136];     // 64 keys x 128 d (+8 pad)   2x17 KB
  __shared__ bf16_t Vs[2][128][72];     // 128 d x 64 keys (+8 pad)   2x18 KB

  const int tid  = threadIdx.x;
  const int lane = tid & 31;
  const int wave = tid >> 5;
  const int n16  = lane & 15;
  const int hi   = (lane & 16) ? 8 : 0;

  const int bh = blockIdx.y;          // b*HEADS + h
  const int bi = bh >> 4;
  const int h  = bh & (HEADS - 1);
  const int q0b = blockIdx.x * 64;    // block q tile
  const int q0w = q0b + wave * 16;    // this wave's 16 q rows

  const float scale = 0.088388347648318447f;  // 1/sqrt(128)

  const bf16_t* Kbase = Kb + ((size_t)bi * SEQ) * DMODEL + h * DHEAD;  // uniform
  const bf16_t* Vbase = Vt + ((size_t)bh * DHEAD) * SEQ;               // uniform

  // cooperative K/V tile staging (128 threads, 8 x 16B chunks each)
  auto stage_kv = [&](int buf, int k0) {
#pragma unroll
    for (int i = 0; i < 8; ++i) {        // K: 64 rows x 16 chunks
      int c = tid + 128 * i;
      int row = c >> 4, off = (c & 15) * 8;
      cp16(&Ks[buf][row][off], Kbase + (size_t)(k0 + row) * DMODEL + off);
    }
#pragma unroll
    for (int i = 0; i < 8; ++i) {        // V: 128 rows x 8 chunks
      int c = tid + 128 * i;
      int row = c >> 3, off = (c & 7) * 8;
      cp16(&Vs[buf][row][off], Vbase + (size_t)row * SEQ + k0 + off);
    }
  };

  // Stage this wave's Q A-fragments into LDS (wave-private, in-order DS)
#pragma unroll
  for (int c = 0; c < 4; ++c) {
    FragU f;
    f.v = load_frag_a(Q + ((size_t)(bi * SEQ + q0w)) * DMODEL + h * DHEAD + c * 32, DMODEL);
    Qs[wave][c][0][lane] = f.u[0];
    Qs[wave][c][1][lane] = f.u[1];
  }

  const v8f vz = {0.f, 0.f, 0.f, 0.f, 0.f, 0.f, 0.f, 0.f};
  v8f o[8];
#pragma unroll
  for (int t = 0; t < 8; ++t) o[t] = vz;
  float rmax[8], rsum[8];
#pragma unroll
  for (int r = 0; r < 8; ++r) { rmax[r] = -INFINITY; rsum[r] = 0.f; }

  const int last_kt = blockIdx.x;  // causal: keys up to q0b+63

  stage_kv(0, 0);
  cp_wait_all();
  __syncthreads();

  for (int kt = 0; kt <= last_kt; ++kt) {
    const int k0 = kt * 64;
    const int buf = kt & 1;
    if (kt < last_kt) stage_kv(buf ^ 1, k0 + 64);  // async prefetch next tile

    // ---- scores: S = Q * K^T (16 WMMAs; one Q chunk live at a time) ----
    v8f s[4];
#pragma unroll
    for (int j = 0; j < 4; ++j) s[j] = vz;
#pragma unroll
    for (int c = 0; c < 4; ++c) {
      FragU qf;
      qf.u[0] = Qs[wave][c][0][lane];
      qf.u[1] = Qs[wave][c][1][lane];
#pragma unroll
      for (int j = 0; j < 4; ++j) {
        v16bf kf = load_frag_b(&Ks[buf][j * 16][c * 32], 136);
        s[j] = WMMA_BF16(qf.v, kf, s[j]);
      }
    }

    // ---- scale + causal mask (diagonal tile only) ----
    const bool diag = (kt == last_kt);
#pragma unroll
    for (int j = 0; j < 4; ++j) {
#pragma unroll
      for (int r = 0; r < 8; ++r) {
        float v = s[j][r] * scale;
        if (diag && (k0 + j * 16 + n16) > (q0w + r + hi)) v = -INFINITY;
        s[j][r] = v;
      }
    }

    // ---- online softmax: row max/sum via 16-lane butterflies ----
    float al[8];
#pragma unroll
    for (int r = 0; r < 8; ++r) {
      float m = fmaxf(fmaxf(s[0][r], s[1][r]), fmaxf(s[2][r], s[3][r]));
      m = fmaxf(m, __shfl_xor(m, 1, 16));
      m = fmaxf(m, __shfl_xor(m, 2, 16));
      m = fmaxf(m, __shfl_xor(m, 4, 16));
      m = fmaxf(m, __shfl_xor(m, 8, 16));
      const float nm = fmaxf(rmax[r], m);
      al[r] = __expf(rmax[r] - nm);
      rmax[r] = nm;

      float sum = 0.f;
#pragma unroll
      for (int j = 0; j < 4; ++j) {
        const float pv = __expf(s[j][r] - nm);
        sum += pv;
        Pl[wave][r + hi][j * 16 + n16] = (bf16_t)pv;
      }
      sum += __shfl_xor(sum, 1, 16);
      sum += __shfl_xor(sum, 2, 16);
      sum += __shfl_xor(sum, 4, 16);
      sum += __shfl_xor(sum, 8, 16);
      rsum[r] = rsum[r] * al[r] + sum;
    }
#pragma unroll
    for (int t = 0; t < 8; ++t)
#pragma unroll
      for (int r = 0; r < 8; ++r) o[t][r] *= al[r];

    // ---- O += P * V (16 WMMAs); P is wave-private, DS ops in-order ----
    v16bf pa0 = load_frag_a(&Pl[wave][0][0], 72);
    v16bf pa1 = load_frag_a(&Pl[wave][0][32], 72);
#pragma unroll
    for (int t = 0; t < 8; ++t) {
      v16bf vb0 = load_frag_b(&Vs[buf][t * 16][0], 72);
      v16bf vb1 = load_frag_b(&Vs[buf][t * 16][32], 72);
      o[t] = WMMA_BF16(pa0, vb0, o[t]);
      o[t] = WMMA_BF16(pa1, vb1, o[t]);
    }

    cp_wait_all();    // next tile landed
    __syncthreads();  // everyone done with this buffer
  }

  // ---- normalize + store attn (bf16, [b][s][h*128+d]) ----
  float rinv[8];
#pragma unroll
  for (int r = 0; r < 8; ++r) rinv[r] = 1.0f / rsum[r];
#pragma unroll
  for (int t = 0; t < 8; ++t)
#pragma unroll
    for (int r = 0; r < 8; ++r)
      attn[((size_t)(bi * SEQ + q0w + r + hi)) * DMODEL + h * DHEAD + t * 16 + n16] =
          (bf16_t)(o[t][r] * rinv[r]);
}

// ---------------------------------------------------------------------------
// Host side
// ---------------------------------------------------------------------------
extern "C" void kernel_launch(void* const* d_in, const int* in_sizes, int n_in,
                              void* d_out, int out_size, void* d_ws, size_t ws_size,
                              hipStream_t stream) {
  const float* x  = (const float*)d_in[0];
  const float* wq = (const float*)d_in[1];
  const float* wk = (const float*)d_in[2];
  const float* wv = (const float*)d_in[3];
  const float* wo = (const float*)d_in[4];

  const size_t SZ_X = (size_t)BATCH * SEQ * DMODEL * sizeof(bf16_t);  // 32 MB
  const size_t SZ_W = (size_t)DMODEL * DMODEL * sizeof(bf16_t);       //  8 MB

  char* w = (char*)d_ws;
  bf16_t* xb   = (bf16_t*)(w);                       // also reused as attn
  bf16_t* wqT  = (bf16_t*)(w + SZ_X);
  bf16_t* wkT  = (bf16_t*)(w + SZ_X + SZ_W);
  bf16_t* wvT  = (bf16_t*)(w + SZ_X + 2 * SZ_W);
  bf16_t* woT  = (bf16_t*)(w + SZ_X + 3 * SZ_W);
  bf16_t* Qb   = (bf16_t*)(w + SZ_X + 4 * SZ_W);
  bf16_t* Kbuf = (bf16_t*)(w + 2 * SZ_X + 4 * SZ_W);
  bf16_t* Vt   = (bf16_t*)(w + 3 * SZ_X + 4 * SZ_W);
  bf16_t* attn = xb;  // x no longer needed after projections

  const int nx = in_sizes[0];  // 16,777,216
  pack_bf16_kernel<<<nx / 1024, 256, 0, stream>>>(x, xb, nx);

  dim3 tb(32, 8), tg(DMODEL / 32, DMODEL / 32);
  transpose_to_bf16_kernel<<<tg, tb, 0, stream>>>(wq, wqT);
  transpose_to_bf16_kernel<<<tg, tb, 0, stream>>>(wk, wkT);
  transpose_to_bf16_kernel<<<tg, tb, 0, stream>>>(wv, wvT);
  transpose_to_bf16_kernel<<<tg, tb, 0, stream>>>(wo, woT);

  dim3 gg(DMODEL / 64, (BATCH * SEQ) / 128);  // (32, 64)
  gemm_bf16_kernel<0><<<gg, 256, 0, stream>>>(xb, wqT, (void*)Qb);
  gemm_bf16_kernel<0><<<gg, 256, 0, stream>>>(xb, wkT, (void*)Kbuf);
  gemm_bf16_kernel<1><<<gg, 256, 0, stream>>>(xb, wvT, (void*)Vt);

  dim3 ag(SEQ / 64, BATCH * HEADS);  // (32, 64)
  flash_attn_kernel<<<ag, 128, 0, stream>>>(Qb, Kbuf, Vt, attn);

  gemm_bf16_kernel<2><<<gg, 256, 0, stream>>>(attn, woT, d_out);
}
